// SimAttention_44959717655056
// MI455X (gfx1250) — matrix operand order repaired and checked by reading, not compiled, not run on hardware
//
#include <hip/hip_runtime.h>

// ---------------- types ----------------
typedef __attribute__((ext_vector_type(16))) __bf16        bf16x16;
typedef __attribute__((ext_vector_type(8)))  float         floatx8;
typedef __attribute__((ext_vector_type(4)))  unsigned int  u32x4;

union Frag {
  u32x4   u4[2];   // 32 bytes = 16 bf16
  bf16x16 bf;
};

#define T_SEQ 2048
#define C_EMB 4096
#define N_H   32
#define N_KV  8
#define H_D   128
#define KV_C  1024
#define NEG_MASK (-40000.0f)

__device__ __forceinline__ unsigned short f2bf(float f) {
  unsigned int u = __float_as_uint(f);
  u += 0x7FFFu + ((u >> 16) & 1u);           // round-to-nearest-even
  return (unsigned short)(u >> 16);
}

__device__ __forceinline__ floatx8 wmma_bf16(const Frag& a, const Frag& b, floatx8 c) {
  return __builtin_amdgcn_wmma_f32_16x16x32_bf16(false, a.bf, false, b.bf,
                                                 (short)0, c, false, false);
}

// ---- CDNA5 async global->LDS copy (ASYNCcnt-tracked DMA, no VGPR round-trip) ----
// GV mode: LDS address from the VDST vgpr, global address from the 64-bit VADDR vgpr pair. 16B per lane.
__device__ __forceinline__ void async_g2l_b128(unsigned lds_byte_off, const void* g) {
  asm volatile("global_load_async_to_lds_b128 %0, %1, off"
               :: "v"(lds_byte_off), "v"(g) : "memory");
}
__device__ __forceinline__ void wait_async0() {
  asm volatile("s_wait_asynccnt 0x0" ::: "memory");
}
// flat address of a __shared__ object: low 32 bits are the LDS offset
__device__ __forceinline__ unsigned lds_off(const void* p) {
  return (unsigned)(size_t)p;
}

// ---------------- elementwise converts ----------------
__global__ __launch_bounds__(256)
void f32_to_bf16_kernel(const float* __restrict__ in, unsigned short* __restrict__ out, int n) {
  int i = blockIdx.x * blockDim.x + threadIdx.x;
  if (i < n) out[i] = f2bf(in[i]);
}

// in: K x N (row-major) f32  ->  out: N x K (row-major) bf16
__global__ __launch_bounds__(256)
void convT_kernel(const float* __restrict__ in, unsigned short* __restrict__ out, int K, int N) {
  int i = blockIdx.x * blockDim.x + threadIdx.x;
  if (i >= K * N) return;
  int k = i / N, n = i - k * N;              // reads coalesced over n
  out[(size_t)n * K + k] = f2bf(in[i]);
}

// ---------------- bf16 GEMM: C(MxN,f32) = A(MxK,bf16) * Bt(NxK,bf16)^T ----------------
// Double-buffered LDS, async DMA staging overlapped with WMMA.
__global__ __launch_bounds__(256)
void gemm_bf16_kernel(const unsigned short* __restrict__ A,
                      const unsigned short* __restrict__ Bt,
                      float* __restrict__ C, int M, int N, int K) {
  __shared__ __attribute__((aligned(16))) unsigned short sA[2][128 * 32];
  __shared__ __attribute__((aligned(16))) unsigned short sB[2][128 * 32];

  const int tid  = threadIdx.x;
  const int lane = tid & 31;
  const int wave = tid >> 5;                 // 0..7
  const int wm   = wave >> 1;                // 0..3  -> 32 rows each
  const int wn   = wave & 1;                 // 0..1  -> 64 cols each
  const int row0 = blockIdx.y * 128;
  const int col0 = blockIdx.x * 128;

  floatx8 zero8 = {0.f,0.f,0.f,0.f,0.f,0.f,0.f,0.f};
  floatx8 acc[2][4];
#pragma unroll
  for (int mt = 0; mt < 2; ++mt)
#pragma unroll
    for (int nt = 0; nt < 4; ++nt) acc[mt][nt] = zero8;

  // staging assignment: thread -> (tile row, 16-short half of 32-wide K slab)
  const int lr = tid >> 1;                   // 0..127
  const int lh = tid & 1;
  const unsigned short* gA = &A [(size_t)(row0 + lr) * K + lh * 16];
  const unsigned short* gB = &Bt[(size_t)(col0 + lr) * K + lh * 16];
  const unsigned lA0 = lds_off(&sA[0][lr * 32 + lh * 16]);
  const unsigned lB0 = lds_off(&sB[0][lr * 32 + lh * 16]);
  const unsigned BUFB = 128 * 32 * 2;        // bytes per buffer

  const int nk = K >> 5;

  // prologue: DMA tile 0 into buffer 0
  {
    async_g2l_b128(lA0,      gA);
    async_g2l_b128(lA0 + 16, gA + 8);
    async_g2l_b128(lB0,      gB);
    async_g2l_b128(lB0 + 16, gB + 8);
  }

  for (int kt = 0; kt < nk; ++kt) {
    const int buf = kt & 1;
    wait_async0();            // this wave's DMA done
    __syncthreads();          // everyone's DMA done -> tile visible

    // prefetch next K-slab into the other buffer while we do WMMA on this one
    if (kt + 1 < nk) {
      const unsigned short* a = gA + (size_t)(kt + 1) * 32;
      const unsigned short* b = gB + (size_t)(kt + 1) * 32;
      const unsigned la = lA0 + (unsigned)((buf ^ 1) * BUFB);
      const unsigned lb = lB0 + (unsigned)((buf ^ 1) * BUFB);
      async_g2l_b128(la,      a);
      async_g2l_b128(la + 16, a + 8);
      async_g2l_b128(lb,      b);
      async_g2l_b128(lb + 16, b + 8);
    }

    const unsigned short* cA = sA[buf];
    const unsigned short* cB = sB[buf];
    Frag a[2], b[4];
    const int kba = (lane < 16) ? 0 : 8;     // A layout: K = kba..kba+7, kba+16..kba+23
    const int kbb = (lane < 16) ? 0 : 8;     // B layout: K = kbb..kbb+7, kbb+16..kbb+23
#pragma unroll
    for (int mt = 0; mt < 2; ++mt) {
      int m = wm * 32 + mt * 16 + (lane & 15);
      a[mt].u4[0] = *(const u32x4*)&cA[m * 32 + kba];
      a[mt].u4[1] = *(const u32x4*)&cA[m * 32 + kba + 16];
    }
#pragma unroll
    for (int nt = 0; nt < 4; ++nt) {
      int n = wn * 64 + nt * 16 + (lane & 15);
      b[nt].u4[0] = *(const u32x4*)&cB[n * 32 + kbb];
      b[nt].u4[1] = *(const u32x4*)&cB[n * 32 + kbb + 16];
    }
#pragma unroll
    for (int mt = 0; mt < 2; ++mt)
#pragma unroll
      for (int nt = 0; nt < 4; ++nt)
        acc[mt][nt] = wmma_bf16(a[mt], b[nt], acc[mt][nt]);

    __syncthreads();          // all reads of buf done before it is re-staged (kt+2)
  }

  // C layout: VGPR r -> rows r (lanes 0-15) / r+8 (lanes 16-31), col = lane&15
  const int rofs = (lane >> 4) << 3;
#pragma unroll
  for (int mt = 0; mt < 2; ++mt)
#pragma unroll
    for (int nt = 0; nt < 4; ++nt)
#pragma unroll
      for (int r = 0; r < 8; ++r) {
        int row = row0 + wm * 32 + mt * 16 + r + rofs;
        int col = col0 + wn * 64 + nt * 16 + (lane & 15);
        C[(size_t)row * N + col] = acc[mt][nt][r];
      }
}

// ---------------- RoPE ----------------
__global__ __launch_bounds__(256)
void rope_q_kernel(const float* __restrict__ qf, const float* __restrict__ cosT,
                   const float* __restrict__ sinT, unsigned short* __restrict__ qb) {
  int i = blockIdx.x * blockDim.x + threadIdx.x;   // T*NH*64
  if (i >= T_SEQ * N_H * 64) return;
  int d = i & 63, h = (i >> 6) & (N_H - 1), t = i >> 11;
  float c = cosT[t * H_D + d];
  float s = sinT[t * H_D + d];
  size_t base = (size_t)t * C_EMB + h * H_D;
  float u1 = qf[base + d], u2 = qf[base + d + 64];
  qb[base + d]      = f2bf(u1 * c - u2 * s);
  qb[base + d + 64] = f2bf(u2 * c + u1 * s);
}

__global__ __launch_bounds__(256)
void rope_kv_kernel(const float* __restrict__ kf, const float* __restrict__ vf,
                    const float* __restrict__ cosT, const float* __restrict__ sinT,
                    float* __restrict__ out_k, float* __restrict__ out_v,
                    unsigned short* __restrict__ kbq, unsigned short* __restrict__ vbq) {
  int i = blockIdx.x * blockDim.x + threadIdx.x;   // T*NKV*64
  if (i >= T_SEQ * N_KV * 64) return;
  int d = i & 63, h = (i >> 6) & (N_KV - 1), t = i >> 9;
  float c = cosT[t * H_D + d];
  float s = sinT[t * H_D + d];
  size_t ib = (size_t)t * KV_C + h * H_D;
  size_t ob = ((size_t)h * T_SEQ + t) * H_D;
  float k1 = kf[ib + d], k2 = kf[ib + d + 64];
  float o1 = k1 * c - k2 * s, o2 = k2 * c + k1 * s;
  out_k[ob + d] = o1;        out_k[ob + d + 64] = o2;
  kbq[ob + d] = f2bf(o1);    kbq[ob + d + 64] = f2bf(o2);
  float v1 = vf[ib + d], v2 = vf[ib + d + 64];
  out_v[ob + d] = v1;        out_v[ob + d + 64] = v2;
  vbq[ob + d] = f2bf(v1);    vbq[ob + d + 64] = f2bf(v2);
}

// ---------------- flash attention (bf16 WMMA, f32 online softmax) ----------------
__global__ __launch_bounds__(128)
void attn_kernel(const unsigned short* __restrict__ qb,   // (T, C) bf16, RoPE'd
                 const unsigned short* __restrict__ kb,   // (NKV, T, HD) bf16 RoPE'd
                 const unsigned short* __restrict__ vb,   // (NKV, T, HD) bf16
                 unsigned short* __restrict__ yout) {     // (T, C) bf16
  __shared__ __attribute__((aligned(16))) unsigned short sK [32 * H_D];  // (key, d)
  __shared__ __attribute__((aligned(16))) unsigned short sVt[H_D * 32];  // (d, key)
  __shared__ __attribute__((aligned(16))) unsigned short sP [4][16 * 32];

  const int tid  = threadIdx.x;
  const int lane = tid & 31;
  const int wave = tid >> 5;              // 0..3
  const int h    = blockIdx.x;
  const int q0   = blockIdx.y * 64;
  const int kvh  = h >> 2;                // H / KVH = 4
  const int qrow0 = q0 + wave * 16;
  const int rofs  = (lane >> 4) << 3;

  // Q fragments: 16 rows x 128 d  ->  4 frags of 16x32, pinned in VGPRs
  Frag aq[4];
  {
    size_t base = (size_t)(qrow0 + (lane & 15)) * C_EMB + h * H_D;
    int kbq = (lane < 16) ? 0 : 8;
#pragma unroll
    for (int c4 = 0; c4 < 4; ++c4) {
      aq[c4].u4[0] = *(const u32x4*)&qb[base + c4 * 32 + kbq];
      aq[c4].u4[1] = *(const u32x4*)&qb[base + c4 * 32 + kbq + 16];
    }
  }

  floatx8 zero8 = {0.f,0.f,0.f,0.f,0.f,0.f,0.f,0.f};
  floatx8 acc[8];
#pragma unroll
  for (int i = 0; i < 8; ++i) acc[i] = zero8;
  float mrow[8], lrow[8];
#pragma unroll
  for (int r = 0; r < 8; ++r) { mrow[r] = -1e30f; lrow[r] = 0.f; }

  const unsigned short* kbase = kb + (size_t)kvh * T_SEQ * H_D;
  const unsigned short* vbase = vb + (size_t)kvh * T_SEQ * H_D;
  unsigned short* sPw = sP[wave];

  const int key = tid >> 2, seg = tid & 3;     // staging: 32 shorts per thread
  const unsigned lK = lds_off(&sK[key * H_D + seg * 32]);

  const int kend = q0 + 64;
  for (int kk = 0; kk < kend; kk += 32) {
    // stage K (32x128) via async DMA; V^T (128x32) via manual transpose
    {
      const unsigned short* gk = &kbase[(size_t)(kk + key) * H_D + seg * 32];
      async_g2l_b128(lK,      gk);
      async_g2l_b128(lK + 16, gk + 8);
      async_g2l_b128(lK + 32, gk + 16);
      async_g2l_b128(lK + 48, gk + 24);
      const unsigned short* gv = &vbase[(size_t)(kk + key) * H_D + seg * 32];
#pragma unroll
      for (int i = 0; i < 32; ++i) sVt[(seg * 32 + i) * 32 + key] = gv[i];
    }
    wait_async0();
    __syncthreads();

    // S = Q @ K^T : two 16x16 tiles over 32 keys
    floatx8 S0 = zero8, S1 = zero8;
    {
      int kb2 = (lane < 16) ? 0 : 8;
      int n0  = lane & 15;
#pragma unroll
      for (int c4 = 0; c4 < 4; ++c4) {
        Frag b0, b1;
        b0.u4[0] = *(const u32x4*)&sK[n0 * H_D + c4 * 32 + kb2];
        b0.u4[1] = *(const u32x4*)&sK[n0 * H_D + c4 * 32 + kb2 + 16];
        b1.u4[0] = *(const u32x4*)&sK[(16 + n0) * H_D + c4 * 32 + kb2];
        b1.u4[1] = *(const u32x4*)&sK[(16 + n0) * H_D + c4 * 32 + kb2 + 16];
        S0 = wmma_bf16(aq[c4], b0, S0);
        S1 = wmma_bf16(aq[c4], b1, S1);
      }
    }

    // causal mask + online softmax (row = r + rofs within wave's 16 rows)
    const int col = kk + (lane & 15);
#pragma unroll
    for (int r = 0; r < 8; ++r) {
      int row = qrow0 + r + rofs;
      float s0 = S0[r] + ((col      > row) ? NEG_MASK : 0.f);
      float s1 = S1[r] + ((col + 16 > row) ? NEG_MASK : 0.f);
      float mx = fmaxf(s0, s1);
#pragma unroll
      for (int off = 8; off >= 1; off >>= 1)
        mx = fmaxf(mx, __shfl_xor(mx, off, 32));      // stays inside 16-lane half
      float mnew  = fmaxf(mrow[r], mx);
      float scale = __expf(mrow[r] - mnew);
      float p0 = __expf(s0 - mnew);
      float p1 = __expf(s1 - mnew);
      float rs = p0 + p1;
#pragma unroll
      for (int off = 8; off >= 1; off >>= 1)
        rs += __shfl_xor(rs, off, 32);
      lrow[r] = lrow[r] * scale + rs;
      mrow[r] = mnew;
#pragma unroll
      for (int nt = 0; nt < 8; ++nt) acc[nt][r] *= scale;
      int m = r + rofs;
      sPw[m * 32 + (lane & 15)]      = f2bf(p0);
      sPw[m * 32 + 16 + (lane & 15)] = f2bf(p1);
    }

    // acc += P(16x32) @ V(32x128)
    {
      Frag ap;
      int m = lane & 15;
      int kbp = (lane < 16) ? 0 : 8;
      ap.u4[0] = *(const u32x4*)&sPw[m * 32 + kbp];
      ap.u4[1] = *(const u32x4*)&sPw[m * 32 + kbp + 16];
      int kb2 = (lane < 16) ? 0 : 8;
#pragma unroll
      for (int nt = 0; nt < 8; ++nt) {
        Frag bv;
        int d = nt * 16 + (lane & 15);
        bv.u4[0] = *(const u32x4*)&sVt[d * 32 + kb2];
        bv.u4[1] = *(const u32x4*)&sVt[d * 32 + kb2 + 16];
        acc[nt] = wmma_bf16(ap, bv, acc[nt]);
      }
    }
    __syncthreads();
  }

  // epilogue: normalize and store bf16
#pragma unroll
  for (int r = 0; r < 8; ++r) {
    float inv = 1.0f / lrow[r];
    size_t row = (size_t)(qrow0 + r + rofs);
#pragma unroll
    for (int nt = 0; nt < 8; ++nt)
      yout[row * C_EMB + h * H_D + nt * 16 + (lane & 15)] = f2bf(acc[nt][r] * inv);
  }
}

// ---------------- launch ----------------
extern "C" void kernel_launch(void* const* d_in, const int* in_sizes, int n_in,
                              void* d_out, int out_size, void* d_ws, size_t ws_size,
                              hipStream_t stream) {
  const float* x   = (const float*)d_in[0];
  const float* Wq  = (const float*)d_in[1];
  const float* Wk  = (const float*)d_in[2];
  const float* Wv  = (const float*)d_in[3];
  const float* Wo  = (const float*)d_in[4];
  const float* cosT= (const float*)d_in[5];
  const float* sinT= (const float*)d_in[6];
  // d_in[7] = mask: causality is applied analytically in attn_kernel

  float* y     = (float*)d_out;                       // (T, C)
  float* out_k = y + (size_t)T_SEQ * C_EMB;           // (NKV, T, HD)
  float* out_v = out_k + (size_t)N_KV * T_SEQ * H_D;  // (NKV, T, HD)

  char* p = (char*)d_ws;
  unsigned short* xb   = (unsigned short*)p; p += (size_t)T_SEQ * C_EMB * 2;   // 16 MiB
  unsigned short* WqT  = (unsigned short*)p; p += (size_t)C_EMB * C_EMB * 2;   // 32 MiB
  unsigned short* WkT  = (unsigned short*)p; p += (size_t)C_EMB * KV_C  * 2;   //  8 MiB
  unsigned short* WvT  = (unsigned short*)p; p += (size_t)C_EMB * KV_C  * 2;   //  8 MiB
  unsigned short* WoT  = (unsigned short*)p; p += (size_t)C_EMB * C_EMB * 2;   // 32 MiB
  float*          qf   = (float*)p;          p += (size_t)T_SEQ * C_EMB * 4;   // 32 MiB
  float*          kf   = (float*)p;          p += (size_t)T_SEQ * KV_C  * 4;   //  8 MiB
  float*          vf   = (float*)p;          p += (size_t)T_SEQ * KV_C  * 4;   //  8 MiB
  unsigned short* qbr  = (unsigned short*)p; p += (size_t)T_SEQ * C_EMB * 2;   // 16 MiB
  unsigned short* kbr  = (unsigned short*)p; p += (size_t)N_KV * T_SEQ * H_D * 2; // 4 MiB
  unsigned short* vbr  = (unsigned short*)p; p += (size_t)N_KV * T_SEQ * H_D * 2; // 4 MiB
  unsigned short* yattb = (unsigned short*)qf; // alias: qf is dead after rope_q

  const int BT = 256;
  // 1) converts
  {
    int n = T_SEQ * C_EMB;
    f32_to_bf16_kernel<<<(n + BT - 1) / BT, BT, 0, stream>>>(x, xb, n);
  }
  convT_kernel<<<(C_EMB * C_EMB + BT - 1) / BT, BT, 0, stream>>>(Wq, WqT, C_EMB, C_EMB);
  convT_kernel<<<(C_EMB * KV_C  + BT - 1) / BT, BT, 0, stream>>>(Wk, WkT, C_EMB, KV_C);
  convT_kernel<<<(C_EMB * KV_C  + BT - 1) / BT, BT, 0, stream>>>(Wv, WvT, C_EMB, KV_C);
  convT_kernel<<<(C_EMB * C_EMB + BT - 1) / BT, BT, 0, stream>>>(Wo, WoT, C_EMB, C_EMB);

  // 2) projections (bf16 WMMA GEMM, f32 out)
  gemm_bf16_kernel<<<dim3(C_EMB / 128, T_SEQ / 128), 256, 0, stream>>>(xb, WqT, qf, T_SEQ, C_EMB, C_EMB);
  gemm_bf16_kernel<<<dim3(KV_C  / 128, T_SEQ / 128), 256, 0, stream>>>(xb, WkT, kf, T_SEQ, KV_C,  C_EMB);
  gemm_bf16_kernel<<<dim3(KV_C  / 128, T_SEQ / 128), 256, 0, stream>>>(xb, WvT, vf, T_SEQ, KV_C,  C_EMB);

  // 3) RoPE (+ emit out_k / out_v f32 outputs and bf16 copies for attention)
  {
    int n = T_SEQ * N_H * 64;
    rope_q_kernel<<<(n + BT - 1) / BT, BT, 0, stream>>>(qf, cosT, sinT, qbr);
  }
  {
    int n = T_SEQ * N_KV * 64;
    rope_kv_kernel<<<(n + BT - 1) / BT, BT, 0, stream>>>(kf, vf, cosT, sinT,
                                                         out_k, out_v, kbr, vbr);
  }

  // 4) flash attention: grid (heads, query blocks of 64)
  attn_kernel<<<dim3(N_H, T_SEQ / 64), 128, 0, stream>>>(qbr, kbr, vbr, yattb);

  // 5) output projection -> y (f32)
  gemm_bf16_kernel<<<dim3(C_EMB / 128, T_SEQ / 128), 256, 0, stream>>>(yattb, WoT, y, T_SEQ, C_EMB, C_EMB);
}
